// MultiHeadedAtt_74423193305421
// MI455X (gfx1250) — matrix operand order repaired and checked by reading, not compiled, not run on hardware
//
#include <hip/hip_runtime.h>
#include <hip/hip_bf16.h>

// ---------------------------------------------------------------------------
// Multi-headed attention for MI455X (gfx1250), wave32 + WMMA bf16 +
// async global->LDS staging of K/V tiles (ASYNCcnt path).
// B=4, S=2048, H=1024, HEADS=16, Hs=64.
// ---------------------------------------------------------------------------

typedef __attribute__((ext_vector_type(16))) __bf16 v16bf;
typedef __attribute__((ext_vector_type(8)))  float  v8f;
typedef int v4i_vs __attribute__((__vector_size__(4 * sizeof(int))));

#define N_HEADS 16
#define HS      64
#define S_LEN   2048
#define B_DIM   4
#define H_DIM   1024

#define AS1 __attribute__((address_space(1)))
#define AS3 __attribute__((address_space(3)))

#if defined(__AMDGCN__) && __has_builtin(__builtin_amdgcn_global_load_async_to_lds_b128)
#define HAVE_ASYNC_LDS 1
#else
#define HAVE_ASYNC_LDS 0
#endif

__device__ __forceinline__ void wait_asynccnt0() {
#if HAVE_ASYNC_LDS
#if __has_builtin(__builtin_amdgcn_s_wait_asynccnt)
  __builtin_amdgcn_s_wait_asynccnt(0);
#else
  asm volatile("s_wait_asynccnt 0" ::: "memory");
#endif
#endif
}

#define WMMA_BF16(a, b, c) \
  __builtin_amdgcn_wmma_f32_16x16x32_bf16(false, (a), false, (b), (short)0, (c), false, false)

// ----- Fragment loaders (contiguous k order used by this file, wave32) --------------
// A-matrix 16x32 bf16: lane L holds row M=L&15.
//   half = L>>4.  element e:  K = half*16 + e
__device__ __forceinline__ v16bf a_frag_f32(const float* __restrict__ src, int stride,
                                            int m_base, int k_base, int lane) {
  int m = lane & 15, half = lane >> 4;
  const float* row = src + (size_t)(m_base + m) * stride + k_base;
  v16bf a;
#pragma unroll
  for (int e = 0; e < 16; ++e) {
    int kk = half * 16 + e;
    a[e] = (__bf16)row[kk];
  }
  return a;
}

__device__ __forceinline__ v16bf a_frag_bf16(const __bf16* __restrict__ src, int stride,
                                             int m_base, int k_base, int lane) {
  int m = lane & 15, half = lane >> 4;
  const __bf16* row = src + (size_t)(m_base + m) * stride + k_base;
  v16bf a;
#pragma unroll
  for (int e = 0; e < 16; ++e) {
    int kk = half * 16 + e;
    a[e] = row[kk];
  }
  return a;
}

// B-matrix 32x16: lane L holds column N=L&15; kg=L>>4 selects K 0..15 / 16..31.
// "Transposed" source: B[k][n] = src[(n_base+n)*stride + k_base + k]
__device__ __forceinline__ v16bf bT_frag_f32(const float* __restrict__ src, int stride,
                                             int n_base, int k_base, int lane) {
  int n = lane & 15, kg = lane >> 4;
  const float* row = src + (size_t)(n_base + n) * stride + k_base + kg * 16;
  v16bf b;
#pragma unroll
  for (int e = 0; e < 16; ++e) b[e] = (__bf16)row[e];
  return b;
}

__device__ __forceinline__ v16bf bT_frag_bf16(const __bf16* __restrict__ src, int stride,
                                              int n_base, int k_base, int lane) {
  int n = lane & 15, kg = lane >> 4;
  const __bf16* row = src + (size_t)(n_base + n) * stride + k_base + kg * 16;
  v16bf b;
#pragma unroll
  for (int e = 0; e < 16; ++e) b[e] = row[e];
  return b;
}

// Direct (K-major) source: B[k][n] = src[(k_base+k)*stride + n_base + n]
__device__ __forceinline__ v16bf b_frag_bf16(const __bf16* __restrict__ src, int stride,
                                             int k_base, int n_base, int lane) {
  int n = lane & 15, kg = lane >> 4;
  v16bf b;
#pragma unroll
  for (int e = 0; e < 16; ++e)
    b[e] = src[(size_t)(k_base + kg * 16 + e) * stride + n_base + n];
  return b;
}

// ---------------------------------------------------------------------------
// Kernel 1: Y = X @ W^T + bias, f32 in -> bf16 out in per-head layout
// ---------------------------------------------------------------------------
__global__ __launch_bounds__(256) void proj_kernel(const float* __restrict__ X,
                                                   const float* __restrict__ W,
                                                   const float* __restrict__ bias,
                                                   __bf16* __restrict__ out) {
  int lane = threadIdx.x & 31, wid = threadIdx.x >> 5;
  int tile = blockIdx.x * 8 + wid;
  const int NT = H_DIM / 16;
  int tm = tile / NT, tn = tile % NT;

  v8f acc = {};
#pragma unroll 4
  for (int k0 = 0; k0 < H_DIM; k0 += 32) {
    v16bf a = a_frag_f32(X, H_DIM, tm * 16, k0, lane);
    v16bf b = bT_frag_f32(W, H_DIM, tn * 16, k0, lane);
    acc = WMMA_BF16(a, b, acc);
  }

  int n_g  = tn * 16 + (lane & 15);
  float bv = bias[n_g];
  int head = n_g >> 6, d = n_g & 63;
  int rbase = tm * 16 + (lane >> 4) * 8;
#pragma unroll
  for (int r = 0; r < 8; ++r) {
    int m_g = rbase + r;
    int bb = m_g / S_LEN, ss = m_g % S_LEN;
    out[((size_t)(bb * N_HEADS + head) * S_LEN + ss) * HS + d] = (__bf16)(acc[r] + bv);
  }
}

// ---------------------------------------------------------------------------
// Kernel 2: flash attention. One block = one (b,h) x 128 query rows
// (8 waves x 16-row q tiles). K/V tiles (32x64 bf16) staged into LDS via
// GLOBAL_LOAD_ASYNC_TO_LDS_B128 and shared by all 8 waves.
// ---------------------------------------------------------------------------
__global__ __launch_bounds__(256) void attn_kernel(const __bf16* __restrict__ Q,
                                                   const __bf16* __restrict__ K,
                                                   const __bf16* __restrict__ V,
                                                   const float* __restrict__ mask,
                                                   __bf16* __restrict__ ctx) {
  __shared__ __bf16 lds_k[32 * HS];          // 4 KB
  __shared__ __bf16 lds_v[32 * HS];          // 4 KB
  __shared__ __bf16 lds_p[8][16 * 32];       // 8 KB (per-wave P staging)

  int lane = threadIdx.x & 31, wid = threadIdx.x >> 5;
  int t = threadIdx.x;
  const int QB = S_LEN / 128;                // 16 q-blocks per (b,h)
  int bh = blockIdx.x / QB;
  int qblk = blockIdx.x % QB;
  int b = bh / N_HEADS, h = bh % N_HEADS;

  const __bf16* Qh = Q + (size_t)bh * S_LEN * HS;
  const __bf16* Kh = K + (size_t)bh * S_LEN * HS;
  const __bf16* Vh = V + (size_t)bh * S_LEN * HS;
  const float* maskb = mask + (size_t)b * S_LEN * S_LEN;

  int q0 = qblk * 128 + wid * 16;
  v16bf qa0 = a_frag_bf16(Qh, HS, q0, 0, lane);
  v16bf qa1 = a_frag_bf16(Qh, HS, q0, 32, lane);

  float m_i[8], l_i[8];
  v8f o0 = {}, o1 = {}, o2 = {}, o3 = {};
#pragma unroll
  for (int r = 0; r < 8; ++r) { m_i[r] = -__builtin_inff(); l_i[r] = 0.f; }

  int rlo = (lane >> 4) * 8;                 // D-layout row base
  int n   = lane & 15;                       // D-layout column
  const float scale = 0.125f;                // 1/sqrt(64)

  // cooperative-copy coordinates: thread t moves one 16B chunk per matrix
  int crow = t >> 3;                         // 0..31 (key row in tile)
  int ccol = (t & 7) * 8;                    // 0..56 (bf16 elements, 16B chunks)

  for (int key0 = 0; key0 < S_LEN; key0 += 32) {
    __syncthreads();                         // everyone done reading lds_k/lds_v
    const __bf16* gK = Kh + (size_t)(key0 + crow) * HS + ccol;
    const __bf16* gV = Vh + (size_t)(key0 + crow) * HS + ccol;
#if HAVE_ASYNC_LDS
    __builtin_amdgcn_global_load_async_to_lds_b128(
        (AS1 v4i_vs*)gK, (AS3 v4i_vs*)&lds_k[crow * HS + ccol], 0, 0);
    __builtin_amdgcn_global_load_async_to_lds_b128(
        (AS1 v4i_vs*)gV, (AS3 v4i_vs*)&lds_v[crow * HS + ccol], 0, 0);
    wait_asynccnt0();
#else
    *(uint4*)&lds_k[crow * HS + ccol] = *(const uint4*)gK;
    *(uint4*)&lds_v[crow * HS + ccol] = *(const uint4*)gV;
#endif
    __syncthreads();                         // tiles visible to all waves

    if (key0 + 32 < S_LEN) {                 // global_prefetch_b8 next tiles
      __builtin_prefetch(Kh + (size_t)(key0 + 32) * HS, 0, 0);
      __builtin_prefetch(Vh + (size_t)(key0 + 32) * HS, 0, 0);
    }

    // ---- scores: two 16x16 tiles over d=64, K read from LDS ----
    v8f s0 = {}, s1 = {};
    {
      v16bf kb;
      kb = bT_frag_bf16(lds_k, HS, 0,  0,  lane); s0 = WMMA_BF16(qa0, kb, s0);
      kb = bT_frag_bf16(lds_k, HS, 0,  32, lane); s0 = WMMA_BF16(qa1, kb, s0);
      kb = bT_frag_bf16(lds_k, HS, 16, 0,  lane); s1 = WMMA_BF16(qa0, kb, s1);
      kb = bT_frag_bf16(lds_k, HS, 16, 32, lane); s1 = WMMA_BF16(qa1, kb, s1);
    }

    // ---- scale + mask + online softmax (row reduce across 16 lanes) ----
    float p0[8], p1[8];
#pragma unroll
    for (int r = 0; r < 8; ++r) {
      int row = q0 + rlo + r;
      float v0 = s0[r] * scale + maskb[(size_t)row * S_LEN + key0 + n];
      float v1 = s1[r] * scale + maskb[(size_t)row * S_LEN + key0 + 16 + n];
      float mx = fmaxf(v0, v1);
#pragma unroll
      for (int off = 8; off >= 1; off >>= 1) mx = fmaxf(mx, __shfl_xor(mx, off, 16));
      float mnew = fmaxf(m_i[r], mx);
      float e0 = __expf(v0 - mnew), e1 = __expf(v1 - mnew);
      float sum = e0 + e1;
#pragma unroll
      for (int off = 8; off >= 1; off >>= 1) sum += __shfl_xor(sum, off, 16);
      float corr = __expf(m_i[r] - mnew);
      l_i[r] = l_i[r] * corr + sum;
      m_i[r] = mnew;
      o0[r] *= corr; o1[r] *= corr; o2[r] *= corr; o3[r] *= corr;
      p0[r] = e0; p1[r] = e1;
    }

    // ---- stage P (16x32 bf16) through per-wave LDS (in-wave DS ordering) ----
#pragma unroll
    for (int r = 0; r < 8; ++r) {
      lds_p[wid][(rlo + r) * 32 + n]      = (__bf16)p0[r];
      lds_p[wid][(rlo + r) * 32 + 16 + n] = (__bf16)p1[r];
    }
    v16bf pa = a_frag_bf16(&lds_p[wid][0], 32, 0, 0, lane);

    // ---- PV: 4 d-tiles of 16, V read from LDS ----
    v16bf vb;
    vb = b_frag_bf16(lds_v, HS, 0, 0,  lane); o0 = WMMA_BF16(pa, vb, o0);
    vb = b_frag_bf16(lds_v, HS, 0, 16, lane); o1 = WMMA_BF16(pa, vb, o1);
    vb = b_frag_bf16(lds_v, HS, 0, 32, lane); o2 = WMMA_BF16(pa, vb, o2);
    vb = b_frag_bf16(lds_v, HS, 0, 48, lane); o3 = WMMA_BF16(pa, vb, o3);
  }

  // ---- normalize and merge heads: ctx[b, s, h*64 + d] (bf16) ----
#pragma unroll
  for (int r = 0; r < 8; ++r) {
    float inv = 1.0f / l_i[r];
    int row = q0 + rlo + r;
    size_t base = ((size_t)b * S_LEN + row) * H_DIM + (size_t)h * HS;
    ctx[base + 0  + n] = (__bf16)(o0[r] * inv);
    ctx[base + 16 + n] = (__bf16)(o1[r] * inv);
    ctx[base + 32 + n] = (__bf16)(o2[r] * inv);
    ctx[base + 48 + n] = (__bf16)(o3[r] * inv);
  }
}

// ---------------------------------------------------------------------------
// Kernel 3: OUT = CTX(bf16) @ WO^T(f32) + bo, f32 row-major out
// ---------------------------------------------------------------------------
__global__ __launch_bounds__(256) void outproj_kernel(const __bf16* __restrict__ X,
                                                      const float* __restrict__ W,
                                                      const float* __restrict__ bias,
                                                      float* __restrict__ out) {
  int lane = threadIdx.x & 31, wid = threadIdx.x >> 5;
  int tile = blockIdx.x * 8 + wid;
  const int NT = H_DIM / 16;
  int tm = tile / NT, tn = tile % NT;

  v8f acc = {};
#pragma unroll 4
  for (int k0 = 0; k0 < H_DIM; k0 += 32) {
    v16bf a = a_frag_bf16(X, H_DIM, tm * 16, k0, lane);
    v16bf b = bT_frag_f32(W, H_DIM, tn * 16, k0, lane);
    acc = WMMA_BF16(a, b, acc);
  }

  int n_g = tn * 16 + (lane & 15);
  float bv = bias[n_g];
  int rbase = tm * 16 + (lane >> 4) * 8;
#pragma unroll
  for (int r = 0; r < 8; ++r)
    out[(size_t)(rbase + r) * H_DIM + n_g] = acc[r] + bv;
}

// ---------------------------------------------------------------------------
extern "C" void kernel_launch(void* const* d_in, const int* in_sizes, int n_in,
                              void* d_out, int out_size, void* d_ws, size_t ws_size,
                              hipStream_t stream) {
  const float* key_in   = (const float*)d_in[0];
  const float* query_in = (const float*)d_in[1];
  const float* value_in = (const float*)d_in[2];
  const float* mask     = (const float*)d_in[3];
  const float* wq = (const float*)d_in[4];  const float* bq = (const float*)d_in[5];
  const float* wk = (const float*)d_in[6];  const float* bk = (const float*)d_in[7];
  const float* wv = (const float*)d_in[8];  const float* bv = (const float*)d_in[9];
  const float* wo = (const float*)d_in[10]; const float* bo = (const float*)d_in[11];

  // Workspace: Q, K, V (per-head layout) and CTX (merged), all bf16: 4 x 16 MB.
  const size_t elems = (size_t)B_DIM * S_LEN * H_DIM;       // 8,388,608
  char* ws = (char*)d_ws;
  __bf16* Qd = (__bf16*)(ws);
  __bf16* Kd = (__bf16*)(ws + 2 * elems);
  __bf16* Vd = (__bf16*)(ws + 4 * elems);
  __bf16* Cd = (__bf16*)(ws + 6 * elems);

  const int M = B_DIM * S_LEN;                              // 8192
  int proj_blocks = (M / 16) * (H_DIM / 16) / 8;            // 4096
  proj_kernel<<<proj_blocks, 256, 0, stream>>>(query_in, wq, bq, Qd);
  proj_kernel<<<proj_blocks, 256, 0, stream>>>(key_in,   wk, bk, Kd);
  proj_kernel<<<proj_blocks, 256, 0, stream>>>(value_in, wv, bv, Vd);

  int attn_blocks = B_DIM * N_HEADS * (S_LEN / 128);        // 1024
  attn_kernel<<<attn_blocks, 256, 0, stream>>>(Qd, Kd, Vd, mask, Cd);

  outproj_kernel<<<proj_blocks, 256, 0, stream>>>(Cd, wo, bo, (float*)d_out);
}
